// RingFlashAttention_16922171146511
// MI455X (gfx1250) — matrix operand order repaired and checked by reading, not compiled, not run on hardware
//
#include <hip/hip_runtime.h>
#include <hip/hip_bf16.h>

typedef __attribute__((ext_vector_type(16))) _Float16 v16h;
typedef __attribute__((ext_vector_type(2)))  _Float16 v2h;
typedef __attribute__((ext_vector_type(8)))  float    v8f;
typedef __attribute__((ext_vector_type(4)))  float    v4f;
typedef __attribute__((ext_vector_type(8)))  unsigned v8u;

#define H_CONST 16
#define S_LOCAL 1024
#define D_HEAD  64

static __device__ __forceinline__ v8f wmma_f16(v16h a, v16h b, v8f c) {
  // D = A(16x32 f16) * B(32x16 f16) + C(16x16 f32)
  return __builtin_amdgcn_wmma_f32_16x16x32_f16(false, a, false, b, (short)0, c, false, false);
}

// One wave owns 32 queries (two 16-q tiles) and walks KV in tiles of 64
// (four 16-kv subtiles). S^T = K*Q^T keeps each softmax row in a lane column;
// scale*log2e is pre-folded into the f16 Q fragment. O^T = V^T * P^T with P^T
// rebuilt via packed-f16 cross-half shuffles + v_perm_b32 (no LDS). V gathers
// are double-buffered across the softmax so loads overlap WMMA/VALU; the
// CONVERTED fragments (ka/va) are also double-buffered by parity so VALU
// writes never hit registers an in-flight WMMA is reading (no WAR NOPs).
// Accumulator rescale is ballot-gated (skipped once the running max settles).
__global__ __launch_bounds__(64)
void rfa_fwd_kernel(const float* __restrict__ Q,
                    const float* __restrict__ Kg,
                    const float* __restrict__ Vg,
                    const int*  __restrict__ wsp,
                    const int*  __restrict__ rkp,
                    float* __restrict__ O)
{
    const int lane = threadIdx.x & 31;
    const int wave = threadIdx.x >> 5;
    const int col  = lane & 15;        // query column this lane owns (N index)
    const int hs   = (lane >> 4) & 1;  // lane half (0: lanes 0-15, 1: lanes 16-31)
    const int h    = blockIdx.y;
    const int qbase = (blockIdx.x * 2 + wave) * 32;   // 32 queries per wave
    const unsigned psel = hs ? 0x07060302u : 0x05040100u;  // v_perm: pick hi16/lo16

    const int world = wsp[0];
    const int rank  = rkp[0];
    const int q_off = rank * S_LOCAL;
    const int Skv   = world * S_LOCAL;   // multiple of 64 -> tile loads stay in bounds

    const float* Qh = Q  + (size_t)h * S_LOCAL * D_HEAD;
    const float* Kh = Kg + (size_t)h * (size_t)Skv * D_HEAD;
    const float* Vh = Vg + (size_t)h * (size_t)Skv * D_HEAD;
    float*       Oh = O  + (size_t)h * S_LOCAL * D_HEAD;

    const float scl = 0.125f * 1.44269504088896340736f; // (1/sqrt(64)) * log2(e)

    // ---- Q^T B-fragments with scale folded in: B[d][q], K(e)=hs*16+e (+32*db)
    v16h qb[2][2];
    #pragma unroll
    for (int qt = 0; qt < 2; ++qt) {
        const float* qrow = Qh + (size_t)(qbase + qt * 16 + col) * D_HEAD;
        #pragma unroll
        for (int db = 0; db < 2; ++db) {
            const float* p = qrow + db * 32 + hs * 16;
            v4f x0 = *(const v4f*)(p +  0);
            v4f x1 = *(const v4f*)(p +  4);
            v4f x2 = *(const v4f*)(p +  8);
            v4f x3 = *(const v4f*)(p + 12);
            #pragma unroll
            for (int i = 0; i < 4; ++i) {
                qb[qt][db][i]      = (_Float16)(x0[i] * scl);
                qb[qt][db][4 + i]  = (_Float16)(x1[i] * scl);
                qb[qt][db][8 + i]  = (_Float16)(x2[i] * scl);
                qb[qt][db][12 + i] = (_Float16)(x3[i] * scl);
            }
        }
    }

    // O^T accumulators: [qt][dt] 16x16 f32 tiles
    v8f acc[8];
    #pragma unroll
    for (int i = 0; i < 8; ++i) acc[i] = (v8f){};

    float m[2] = { -INFINITY, -INFINITY };
    float l[2] = { 0.0f, 0.0f };
    int   qpos[2];
    qpos[0] = q_off + qbase + col;
    qpos[1] = q_off + qbase + 16 + col;
    const int qpmin0 = q_off + qbase;

    int kv_len = qpmin0 + 32;               // causal: last valid kv is qpmin0+31
    if (kv_len > Skv) kv_len = Skv;

    for (int kb = 0; kb < kv_len; kb += 64) {
        // ---- prefetch next 64-KV tile while this tile's WMMAs run ----
        if (kb + 64 < kv_len) {
            __builtin_prefetch(Kh + (size_t)(kb + 64 + lane) * D_HEAD, 0, 3);
            __builtin_prefetch(Kh + (size_t)(kb + 96 + lane) * D_HEAD, 0, 3);
            __builtin_prefetch(Vh + (size_t)(kb + 64 + lane) * D_HEAD, 0, 3);
            __builtin_prefetch(Vh + (size_t)(kb + 96 + lane) * D_HEAD, 0, 3);
        }

        // ---- S^T tiles: per 16-kv subtile, build K A-frags and fire QK WMMAs.
        // ka fragments alternate register sets by subtile parity so the cvt
        // writes never WAR-collide with the previous subtile's WMMA reads.
        v8f  st[2][4];
        v16h kaP[2][2];    // [parity][dblock]
        #pragma unroll
        for (int t = 0; t < 4; ++t) {
            const int pr = t & 1;
            const float* kr = Kh + (size_t)(kb + t * 16 + col) * D_HEAD + hs * 16;
            #pragma unroll
            for (int db = 0; db < 2; ++db) {
                const float* p = kr + db * 32;
                v4f x0 = *(const v4f*)(p +  0);
                v4f x1 = *(const v4f*)(p +  4);
                v4f x2 = *(const v4f*)(p +  8);
                v4f x3 = *(const v4f*)(p + 12);
                #pragma unroll
                for (int i = 0; i < 4; ++i) {
                    kaP[pr][db][i]      = (_Float16)x0[i];
                    kaP[pr][db][4 + i]  = (_Float16)x1[i];
                    kaP[pr][db][8 + i]  = (_Float16)x2[i];
                    kaP[pr][db][12 + i] = (_Float16)x3[i];
                }
            }
            #pragma unroll
            for (int qt = 0; qt < 2; ++qt) {
                v8f s = {};
                s = wmma_f16(kaP[pr][0], qb[qt][0], s);
                s = wmma_f16(kaP[pr][1], qb[qt][1], s);
                st[qt][t] = s;
            }
        }

        // ---- preload V gather stage 0 (dt=0, hh=0); softmax hides its latency
        const float* vbase = Vh + (size_t)(kb + hs * 16) * D_HEAD + col;
        float rawA[16], rawB[16];
        #pragma unroll
        for (int j = 0; j < 8; ++j) {
            rawA[j]     = vbase[(size_t)j * D_HEAD];
            rawA[8 + j] = vbase[(size_t)(j + 8) * D_HEAD];
        }

        // ---- per q-tile: online softmax over 64 kv + P^T fragments ----
        v16h pB[2][2];     // [qt][kv-half]
        #pragma unroll
        for (int qt = 0; qt < 2; ++qt) {
            const bool needMask = (kb + 63 > qpmin0 + 16 * qt);   // wave-uniform
            if (needMask) {
                #pragma unroll
                for (int t = 0; t < 4; ++t)
                    #pragma unroll
                    for (int r = 0; r < 8; ++r) {
                        const int kva = kb + t * 16 + hs * 8 + r;
                        if (kva > qpos[qt]) st[qt][t][r] = -INFINITY;
                    }
            }
            float mloc = -INFINITY;
            #pragma unroll
            for (int t = 0; t < 4; ++t)
                #pragma unroll
                for (int r = 0; r < 8; ++r) mloc = fmaxf(mloc, st[qt][t][r]);
            mloc = fmaxf(mloc, __shfl_xor(mloc, 16, 32));

            // Ballot-gated rescale: skip entirely (wave-uniform) once the
            // running max has settled; corr==1 for lanes whose max didn't move.
            if (__ballot(mloc > m[qt])) {
                const float mnew = fmaxf(m[qt], mloc);
                const float corr = exp2f(m[qt] - mnew);   // 0 on first tile
                l[qt] *= corr;
                #pragma unroll
                for (int dt = 0; dt < 4; ++dt) acc[qt * 4 + dt] *= corr;
                m[qt] = mnew;
            }

            float ssum = 0.0f;
            #pragma unroll
            for (int t = 0; t < 4; ++t)
                #pragma unroll
                for (int r = 0; r < 8; ++r) {
                    const float p = exp2f(st[qt][t][r] - m[qt]);
                    st[qt][t][r] = p;            // reuse S storage for P
                    ssum += p;
                }
            ssum += __shfl_xor(ssum, 16, 32);
            l[qt] += ssum;

            // P^T B-fragments, one per 32-kv half: pack (pa,pb) to f16 pairs,
            // one cross-half shuffle of the packed word per row, then place
            // hi/lo halves with v_perm_b32 (per-lane byte selector).
            #pragma unroll
            for (int hh = 0; hh < 2; ++hh) {
                unsigned tt[8], uu[8];
                #pragma unroll
                for (int r = 0; r < 8; ++r) {
                    v2h hp;
                    hp[0] = (_Float16)st[qt][2 * hh][r];      // kv rel  hs*8+r
                    hp[1] = (_Float16)st[qt][2 * hh + 1][r];  // kv rel 16+hs*8+r
                    const unsigned pk  = __builtin_bit_cast(unsigned, hp);
                    const unsigned swz = (unsigned)__shfl_xor((int)pk, 16, 32);
                    tt[r] = hs ? swz : pk;   // feeds elements e=0..7
                    uu[r] = hs ? pk : swz;   // feeds elements e=8..15
                }
                v8u iv;
                #pragma unroll
                for (int j = 0; j < 4; ++j) {
                    iv[j]     = __builtin_amdgcn_perm(tt[2 * j + 1], tt[2 * j], psel);
                    iv[4 + j] = __builtin_amdgcn_perm(uu[2 * j + 1], uu[2 * j], psel);
                }
                pB[qt][hh] = __builtin_bit_cast(v16h, iv);
            }
        }

        // ---- O^T += V^T * P^T, software-pipelined: load stage i+1 while
        // converting/WMMA-ing stage i. Raw gathers AND converted va fragments
        // alternate register sets by stage parity (partial loadcnt waits, no
        // WAR hazard NOPs against in-flight WMMAs).
        v16h vaP[2];
        #pragma unroll
        for (int idx = 0; idx < 8; ++idx) {
            const int dt = idx >> 1;
            const int hh = idx & 1;
            const int pr = idx & 1;
            const float* cur = pr ? rawB : rawA;
            if (idx < 7) {
                const int nidx = idx + 1;
                const float* np = vbase + (size_t)((nidx & 1) * 32) * D_HEAD
                                        + (nidx >> 1) * 16;
                float* nxt = pr ? rawA : rawB;
                #pragma unroll
                for (int j = 0; j < 8; ++j) {
                    nxt[j]     = np[(size_t)j * D_HEAD];
                    nxt[8 + j] = np[(size_t)(j + 8) * D_HEAD];
                }
            }
            #pragma unroll
            for (int j = 0; j < 16; ++j) vaP[pr][j] = (_Float16)cur[j];
            acc[dt]     = wmma_f16(vaP[pr], pB[0][hh], acc[dt]);
            acc[4 + dt] = wmma_f16(vaP[pr], pB[1][hh], acc[4 + dt]);
        }
    }

    // ---- epilogue: out[q][d] = acc^T / l ; lane owns q=col, d = dt*16+hs*8+r
    #pragma unroll
    for (int qt = 0; qt < 2; ++qt) {
        const float invl = (l[qt] > 0.0f) ? (1.0f / l[qt]) : 0.0f;
        float* orow = Oh + (size_t)(qbase + qt * 16 + col) * D_HEAD + hs * 8;
        #pragma unroll
        for (int r = 0; r < 8; ++r) {
            orow[r]      = acc[qt * 4 + 0][r] * invl;
            orow[16 + r] = acc[qt * 4 + 1][r] * invl;
            orow[32 + r] = acc[qt * 4 + 2][r] * invl;
            orow[48 + r] = acc[qt * 4 + 3][r] * invl;
        }
    }
}

extern "C" void kernel_launch(void* const* d_in, const int* in_sizes, int n_in,
                              void* d_out, int out_size, void* d_ws, size_t ws_size,
                              hipStream_t stream) {
    const float* q  = (const float*)d_in[0];
    const float* k  = (const float*)d_in[1];
    const float* v  = (const float*)d_in[2];
    const int*   ws = (const int*)d_in[3];   // world_size
    const int*   rk = (const int*)d_in[4];   // rank
    float* out = (float*)d_out;

    // 32 query-groups of 32 rows, 2 waves per block -> grid (16, H)
    dim3 grid(S_LOCAL / 64, H_CONST);
    dim3 block(64);
    rfa_fwd_kernel<<<grid, block, 0, stream>>>(q, k, v, ws, rk, out);
}
